// MOE_40192303956454
// MI455X (gfx1250) — compile-verified, matrix operand order checked by reading, not run on hardware
//
#include <hip/hip_runtime.h>
#include <cstdint>

#define C_EMBD 1184
#define N_EXP  16
#define HID    4736
#define NTOK   8192
#define NROWS  (2 * NTOK)          // total routed (token, expert) rows
#define OUT_ELEMS (NTOK * C_EMBD)

#define TM 64     // rows per block tile
#define TN 128    // cols per block tile
#define TK 64     // K per LDS stage (2 WMMA K-steps of 32)
#define LDSP 72   // padded LDS row stride (halves): 144B = 9*16B keeps 16B alignment

typedef __attribute__((ext_vector_type(16))) _Float16 v16h;
typedef __attribute__((ext_vector_type(8)))  _Float16 v8h;
typedef __attribute__((ext_vector_type(4)))  _Float16 v4h;
typedef __attribute__((ext_vector_type(8)))  float    v8f;
typedef __attribute__((ext_vector_type(4)))  int      v4i;

#if defined(__gfx1250__) && __has_builtin(__builtin_amdgcn_global_load_async_to_lds_b128) && __has_builtin(__builtin_amdgcn_s_wait_asynccnt)
#define USE_ASYNC_LDS 1
#else
#define USE_ASYNC_LDS 0
#endif

__device__ __forceinline__ v16h make_frag(const _Float16* lo, const _Float16* hi) {
  v8h a = *(const v8h*)lo;   // ds_load_b128
  v8h b = *(const v8h*)hi;   // ds_load_b128
  v16h r;
#pragma unroll
  for (int i = 0; i < 8; ++i) { r[i] = a[i]; r[i + 8] = b[i]; }
  return r;
}

__device__ __forceinline__ float fast_rcp(float v) {
#if __has_builtin(__builtin_amdgcn_rcpf)
  return __builtin_amdgcn_rcpf(v);   // v_rcp_f32
#else
  return 1.0f / v;
#endif
}

// ---------------------------------------------------------------- zero
__global__ void zero_kernel(float* __restrict__ out, int* __restrict__ counts) {
  int idx = blockIdx.x * blockDim.x + threadIdx.x;
  if (idx < OUT_ELEMS) out[idx] = 0.0f;
  if (idx < N_EXP) counts[idx] = 0;
}

// ---------------------------------------------------------------- gate + top2
__global__ __launch_bounds__(256) void gate_topk_kernel(
    const float* __restrict__ x, const float* __restrict__ gw,
    const float* __restrict__ gb, int* __restrict__ counts,
    float* __restrict__ topv, int* __restrict__ topi) {
  int wave = blockIdx.x * (blockDim.x >> 5) + (threadIdx.x >> 5);
  int lane = threadIdx.x & 31;
  if (wave >= NTOK) return;
  const float* xr = x + (size_t)wave * C_EMBD;
  float s[N_EXP];
#pragma unroll
  for (int e = 0; e < N_EXP; ++e) s[e] = 0.0f;
  for (int c = lane; c < C_EMBD; c += 32) {   // coalesced across the wave
    float xv = xr[c];
    const float* wr = gw + (size_t)c * N_EXP;
#pragma unroll
    for (int e = 0; e < N_EXP; ++e) s[e] += xv * wr[e];
  }
#pragma unroll
  for (int off = 16; off >= 1; off >>= 1) {
#pragma unroll
    for (int e = 0; e < N_EXP; ++e) s[e] += __shfl_xor(s[e], off, 32);
  }
  if (lane == 0) {
#pragma unroll
    for (int e = 0; e < N_EXP; ++e) s[e] += gb[e];
    int i0 = 0; float v0 = s[0];
    for (int e = 1; e < N_EXP; ++e) if (s[e] > v0) { v0 = s[e]; i0 = e; }
    int i1 = -1; float v1 = -3.4e38f;
    for (int e = 0; e < N_EXP; ++e) if (e != i0 && s[e] > v1) { v1 = s[e]; i1 = e; }
    topv[2 * wave]     = v0; topi[2 * wave]     = i0;
    topv[2 * wave + 1] = v1; topi[2 * wave + 1] = i1;
    atomicAdd(&counts[i0], 1);
    atomicAdd(&counts[i1], 1);
  }
}

// ---------------------------------------------------------------- tiny scan
__global__ void scan_kernel(const int* __restrict__ counts,
                            int* __restrict__ offsets, int* __restrict__ cursor) {
  if (threadIdx.x == 0 && blockIdx.x == 0) {
    int acc = 0;
    for (int e = 0; e < N_EXP; ++e) { offsets[e] = acc; cursor[e] = acc; acc += counts[e]; }
    offsets[N_EXP] = acc;
  }
}

// ---------------------------------------------------------------- scatter rows
__global__ void scatter_kernel(const float* __restrict__ topv,
                               const int* __restrict__ topi,
                               int* __restrict__ cursor,
                               int* __restrict__ tok_list,
                               float* __restrict__ tok_wt) {
  int n = blockIdx.x * blockDim.x + threadIdx.x;
  if (n >= NTOK) return;
#pragma unroll
  for (int k = 0; k < 2; ++k) {
    int e = topi[2 * n + k];
    int pos = atomicAdd(&cursor[e], 1);
    tok_list[pos] = n;
    tok_wt[pos]   = topv[2 * n + k];
  }
}

// ---------------------------------------------------------------- pass A: H = (x@up+b) * silu(x@gproj+b)
__global__ __launch_bounds__(256) void moe_up_gate_kernel(
    const float* __restrict__ x,
    const float* __restrict__ up_w, const float* __restrict__ up_b,
    const float* __restrict__ gp_w, const float* __restrict__ gp_b,
    const int* __restrict__ offsets, const int* __restrict__ tok_list,
    _Float16* __restrict__ H) {
  int e    = blockIdx.y >> 7;
  int tile = blockIdx.y & 127;
  int rbeg = offsets[e], rend = offsets[e + 1];
  int m0 = rbeg + tile * TM;
  if (m0 >= rend) return;                    // static grid, dynamic early-exit
  int mvalid = rend - m0; if (mvalid > TM) mvalid = TM;
  int n0 = blockIdx.x * TN;

  __shared__ __align__(16) _Float16 As[TM][LDSP];
  __shared__ __align__(16) _Float16 Bu[TN][LDSP];
  __shared__ __align__(16) _Float16 Bg[TN][LDSP];
  __shared__ int toks[TM];

  int tid = threadIdx.x;
  for (int i = tid; i < TM; i += 256)
    toks[i] = tok_list[m0 + ((i < mvalid) ? i : 0)];
  __syncthreads();

  int lane = tid & 31;
  int wid  = tid >> 5;                 // 8 waves: 2 (M) x 4 (N)
  int wm = (wid & 1) * 32;
  int wn = (wid >> 1) * 32;
  int ks = (lane & 16) ? 8 : 0;        // K-half select per ISA A/B layout
  int fr = lane & 15;

  v8f zf = {};
  v8f accu[2][2], accg[2][2];
#pragma unroll
  for (int a = 0; a < 2; ++a)
#pragma unroll
    for (int b = 0; b < 2; ++b) { accu[a][b] = zf; accg[a][b] = zf; }

  for (int kk = 0; kk < C_EMBD; kk += TK) {
    // ---- stage A tile (gathered rows): float4 load -> packed v4h (ds_store_b64)
    for (int idx = tid; idx < (TM * TK) / 4; idx += 256) {
      int r = idx >> 4, q = (idx & 15) * 4;
      v4h p;
      if (kk + q < C_EMBD) {
        float4 xv = *(const float4*)&x[(size_t)toks[r] * C_EMBD + kk + q];
        p[0] = (_Float16)xv.x; p[1] = (_Float16)xv.y;
        p[2] = (_Float16)xv.z; p[3] = (_Float16)xv.w;
      } else {
        p[0] = p[1] = p[2] = p[3] = (_Float16)0.f;
      }
      *(v4h*)&As[r][q] = p;
    }
    // ---- stage B tiles (up & gproj), 4(k)x4(n) micro-tile per task:
    //      4x float4 global loads, 4x packed v4h ds_store_b64 (k contiguous in [n][k])
    for (int idx = tid; idx < (TK / 4) * (TN / 4); idx += 256) {   // 512 tasks
      int kq = (idx >> 5) * 4;
      int nq = (idx & 31) * 4;
      float4 wu[4], wg[4];
#pragma unroll
      for (int i = 0; i < 4; ++i) {
        if (kk + kq + i < C_EMBD) {
          size_t wi = ((size_t)e * C_EMBD + kk + kq + i) * HID + n0 + nq;
          wu[i] = *(const float4*)&up_w[wi];
          wg[i] = *(const float4*)&gp_w[wi];
        } else {
          wu[i] = make_float4(0.f, 0.f, 0.f, 0.f);
          wg[i] = make_float4(0.f, 0.f, 0.f, 0.f);
        }
      }
#pragma unroll
      for (int j = 0; j < 4; ++j) {
        v4h pu, pg;
#pragma unroll
        for (int i = 0; i < 4; ++i) {
          pu[i] = (_Float16)((&wu[i].x)[j]);
          pg[i] = (_Float16)((&wg[i].x)[j]);
        }
        *(v4h*)&Bu[nq + j][kq] = pu;
        *(v4h*)&Bg[nq + j][kq] = pg;
      }
    }
    // ---- prefetch next K tile (one 128B line per thread per stream)
    int kkn = kk + TK;
    if (kkn < C_EMBD) {
      int pk = tid >> 2, part = (tid & 3) * 32;              // 64 rows x 4x128B
      size_t wn_i = ((size_t)e * C_EMBD + kkn + pk) * HID + n0 + part;
      __builtin_prefetch(&up_w[wn_i], 0, 1);                 // global_prefetch_b8
      __builtin_prefetch(&gp_w[wn_i], 0, 1);
      if (tid < 128) {
        int r = tid >> 1, p2 = (tid & 1) * 32;
        __builtin_prefetch(&x[(size_t)toks[r] * C_EMBD + kkn + p2], 0, 1);
      }
    }
    __syncthreads();

#pragma unroll
    for (int s = 0; s < TK; s += 32) {   // 2 WMMA K-steps per stage
      v16h afr[2], bfu[2], bfg[2];
#pragma unroll
      for (int a = 0; a < 2; ++a)
        afr[a] = make_frag(&As[wm + 16 * a + fr][s + ks],
                           &As[wm + 16 * a + fr][s + ks + 16]);
#pragma unroll
      for (int b = 0; b < 2; ++b) {
        bfu[b] = make_frag(&Bu[wn + 16 * b + fr][s + ks],
                           &Bu[wn + 16 * b + fr][s + ks + 16]);
        bfg[b] = make_frag(&Bg[wn + 16 * b + fr][s + ks],
                           &Bg[wn + 16 * b + fr][s + ks + 16]);
      }
#pragma unroll
      for (int a = 0; a < 2; ++a)
#pragma unroll
        for (int b = 0; b < 2; ++b) {
          accu[a][b] = __builtin_amdgcn_wmma_f32_16x16x32_f16(
              false, afr[a], false, bfu[b], (short)0, accu[a][b], false, false);
          accg[a][b] = __builtin_amdgcn_wmma_f32_16x16x32_f16(
              false, afr[a], false, bfg[b], (short)0, accg[a][b], false, false);
        }
    }
    __syncthreads();
  }

  int mh = (lane >> 4) << 3;   // C/D layout: M = r + 8*(lane>=16)
#pragma unroll
  for (int a = 0; a < 2; ++a) {
#pragma unroll
    for (int b = 0; b < 2; ++b) {
      int ncol = n0 + wn + 16 * b + fr;
      float ub  = up_b[e * HID + ncol];
      float gbv = gp_b[e * HID + ncol];
#pragma unroll
      for (int r = 0; r < 8; ++r) {
        int lrow = wm + 16 * a + r + mh;
        if (lrow < mvalid) {
          float u = accu[a][b][r] + ub;
          float g = accg[a][b][r] + gbv;
          float h = u * g * fast_rcp(1.0f + __expf(-g));   // u * silu(g), v_rcp_f32
          H[(size_t)(m0 + lrow) * HID + ncol] = (_Float16)h;
        }
      }
    }
  }
}

// ---------------------------------------------------------------- pass B: out += wt * (H @ down + b)
__global__ __launch_bounds__(256) void moe_down_kernel(
    const _Float16* __restrict__ H,
    const float* __restrict__ dw, const float* __restrict__ db,
    const int* __restrict__ offsets, const int* __restrict__ tok_list,
    const float* __restrict__ tok_wt, float* __restrict__ out) {
  int e    = blockIdx.y >> 7;
  int tile = blockIdx.y & 127;
  int rbeg = offsets[e], rend = offsets[e + 1];
  int m0 = rbeg + tile * TM;
  if (m0 >= rend) return;
  int mvalid = rend - m0; if (mvalid > TM) mvalid = TM;
  int c0 = blockIdx.x * TN;

  __shared__ __align__(16) _Float16 As[TM][LDSP];
  __shared__ __align__(16) _Float16 Bd[TN][LDSP];
  __shared__ int   toks[TM];
  __shared__ float wts[TM];

  int tid = threadIdx.x;
  for (int i = tid; i < TM; i += 256) {
    int src = m0 + ((i < mvalid) ? i : 0);
    toks[i] = tok_list[src];
    wts[i]  = tok_wt[src];
  }

  int lane = tid & 31;
  int wid  = tid >> 5;
  int wm = (wid & 1) * 32;
  int wn = (wid >> 1) * 32;
  int ks = (lane & 16) ? 8 : 0;
  int fr = lane & 15;

  v8f zf = {};
  v8f acc[2][2];
#pragma unroll
  for (int a = 0; a < 2; ++a)
#pragma unroll
    for (int b = 0; b < 2; ++b) acc[a][b] = zf;

  for (int kk = 0; kk < HID; kk += TK) {   // HID % 64 == 0, no K guard needed
    // ---- A tile from f16 H: pure byte copy -> async DMA to LDS when available
    for (int idx = tid; idx < (TM * TK) / 8; idx += 256) {   // 512 x b128
      int r = idx >> 3, q = (idx & 7) * 8;
      const _Float16* src = &H[(size_t)(m0 + r) * HID + kk + q];
      _Float16* dst = &As[r][q];
#if USE_ASYNC_LDS
      __builtin_amdgcn_global_load_async_to_lds_b128((v4i*)src, (v4i*)dst, 0, 0);
#else
      *(v8h*)dst = *(const v8h*)src;
#endif
    }
    // ---- B tile (down_w) 4(k)x4(n) micro-tile, packed v4h stores, col guard
    for (int idx = tid; idx < (TK / 4) * (TN / 4); idx += 256) {
      int kq = (idx >> 5) * 4;
      int nq = (idx & 31) * 4;
      int c  = c0 + nq;
      float4 wv[4];
#pragma unroll
      for (int i = 0; i < 4; ++i) {
        if (c < C_EMBD)
          wv[i] = *(const float4*)&dw[((size_t)e * HID + kk + kq + i) * C_EMBD + c];
        else
          wv[i] = make_float4(0.f, 0.f, 0.f, 0.f);
      }
#pragma unroll
      for (int j = 0; j < 4; ++j) {
        v4h pd;
#pragma unroll
        for (int i = 0; i < 4; ++i) pd[i] = (_Float16)((&wv[i].x)[j]);
        *(v4h*)&Bd[nq + j][kq] = pd;
      }
    }
    // ---- prefetch next K tile
    int kkn = kk + TK;
    if (kkn < HID) {
      int pk = tid >> 2, part = (tid & 3) * 32;
      __builtin_prefetch(&dw[((size_t)e * HID + kkn + pk) * C_EMBD + c0 + part], 0, 1);
      if (tid < TM)
        __builtin_prefetch(&H[(size_t)(m0 + tid) * HID + kkn], 0, 1);
    }
#if USE_ASYNC_LDS
    __builtin_amdgcn_s_wait_asynccnt(0);   // async LDS writes landed
#endif
    __syncthreads();

#pragma unroll
    for (int s = 0; s < TK; s += 32) {
      v16h afr[2], bfd[2];
#pragma unroll
      for (int a = 0; a < 2; ++a)
        afr[a] = make_frag(&As[wm + 16 * a + fr][s + ks],
                           &As[wm + 16 * a + fr][s + ks + 16]);
#pragma unroll
      for (int b = 0; b < 2; ++b)
        bfd[b] = make_frag(&Bd[wn + 16 * b + fr][s + ks],
                           &Bd[wn + 16 * b + fr][s + ks + 16]);
#pragma unroll
      for (int a = 0; a < 2; ++a)
#pragma unroll
        for (int b = 0; b < 2; ++b)
          acc[a][b] = __builtin_amdgcn_wmma_f32_16x16x32_f16(
              false, afr[a], false, bfd[b], (short)0, acc[a][b], false, false);
    }
    __syncthreads();
  }

  int mh = (lane >> 4) << 3;
#pragma unroll
  for (int a = 0; a < 2; ++a) {
#pragma unroll
    for (int b = 0; b < 2; ++b) {
      int c = c0 + wn + 16 * b + fr;
      if (c < C_EMBD) {
        float dbv = db[e * C_EMBD + c];
#pragma unroll
        for (int r = 0; r < 8; ++r) {
          int lrow = wm + 16 * a + r + mh;
          if (lrow < mvalid) {
            float v = (acc[a][b][r] + dbv) * wts[lrow];
            atomicAdd(&out[(size_t)toks[lrow] * C_EMBD + c], v);  // exactly 2 adds/elem
          }
        }
      }
    }
  }
}

// ---------------------------------------------------------------- launch
extern "C" void kernel_launch(void* const* d_in, const int* in_sizes, int n_in,
                              void* d_out, int out_size, void* d_ws, size_t ws_size,
                              hipStream_t stream) {
  const float* x   = (const float*)d_in[0];
  const float* gw  = (const float*)d_in[1];
  const float* gb  = (const float*)d_in[2];
  const float* upw = (const float*)d_in[3];
  const float* upb = (const float*)d_in[4];
  const float* gpw = (const float*)d_in[5];
  const float* gpb = (const float*)d_in[6];
  const float* dww = (const float*)d_in[7];
  const float* dwb = (const float*)d_in[8];
  float* out = (float*)d_out;

  // workspace layout
  char* w = (char*)d_ws;
  int*   counts   = (int*)w;                    // 16
  int*   offsets  = counts + 16;                // 17 (padded to 32)
  int*   cursor   = offsets + 32;               // 16
  int*   tok_list = cursor + 16;                // NROWS
  int*   topi     = tok_list + NROWS;           // NROWS
  float* tok_wt   = (float*)(topi + NROWS);     // NROWS
  float* topv     = tok_wt + NROWS;             // NROWS
  uintptr_t hoff  = (uintptr_t)(topv + NROWS);
  hoff = (hoff + 255) & ~(uintptr_t)255;
  _Float16* H = (_Float16*)hoff;                // (NROWS + 64) * HID halves (tile over-read pad)

  (void)in_sizes; (void)n_in; (void)out_size; (void)ws_size;

  zero_kernel<<<(OUT_ELEMS + 255) / 256, 256, 0, stream>>>(out, counts);
  gate_topk_kernel<<<NTOK / 8, 256, 0, stream>>>(x, gw, gb, counts, topv, topi);
  scan_kernel<<<1, 1, 0, stream>>>(counts, offsets, cursor);
  scatter_kernel<<<NTOK / 256, 256, 0, stream>>>(topv, topi, cursor, tok_list, tok_wt);

  dim3 gridA(HID / TN, N_EXP * 128);                     // 37 x 2048
  moe_up_gate_kernel<<<gridA, 256, 0, stream>>>(x, upw, upb, gpw, gpb,
                                                offsets, tok_list, H);
  dim3 gridB((C_EMBD + TN - 1) / TN, N_EXP * 128);       // 10 x 2048
  moe_down_kernel<<<gridB, 256, 0, stream>>>(H, dww, dwb, offsets,
                                             tok_list, tok_wt, out);
}
